// NeighbourSupport_44006234915177
// MI455X (gfx1250) — matrix operand order repaired and checked by reading, not compiled, not run on hardware
//
#include <hip/hip_runtime.h>
#include <hip/hip_bf16.h>

typedef __attribute__((ext_vector_type(16))) __bf16 v16bf;
typedef __attribute__((ext_vector_type(8)))  float  v8f;

#define BATCH 8
#define CH    256
#define CKC   32
#define CVC   32
#define HH    128
#define WW    128
#define TS    16
#define HS    18              // haloed tile: 18x18
#define NPIX  (HS*HS)         // 324
#define NMT   ((NPIX+15)/16)  // 21 M-tiles over the haloed region
#define NROW  (NMT*16)        // 336: padded so D-stores need no guard

// A-matrix (16x32 bf16): K index for VGPR-pair j, lane-half h (ISA 7.12.2)
__device__ __forceinline__ int kmapA(int h, int j) {
    return (j < 4) ? (h*8 + 2*j) : (16 + h*8 + 2*(j-4));
}
// B-matrix (32x16 bf16): K index for VGPR-pair j, lane-half h
__device__ __forceinline__ int kmapB(int h, int j) {
    return (j < 4) ? (h*8 + 2*j) : (16 + h*8 + 2*(j-4));
}
// inverse of kmapA: channel c -> (lane half h, element index e in 16-wide frag)
__device__ __forceinline__ void invA(int c, int& h, int& e) {
    int j;
    if (c < 8)       { h = 0; j = c >> 1; }
    else if (c < 16) { h = 1; j = (c - 8) >> 1; }
    else if (c < 24) { h = 0; j = 4 + ((c - 16) >> 1); }
    else             { h = 1; j = 4 + ((c - 24) >> 1); }
    e = 2*j + (c & 1);
}

__global__ __launch_bounds__(256)
void ns_fused(const float* __restrict__ x,
              const float* __restrict__ k1_w, const float* __restrict__ k1_b,
              const float* __restrict__ dw_w, const float* __restrict__ dw_b,
              const float* __restrict__ k3_w, const float* __restrict__ k3_b,
              const float* __restrict__ v_w,  const float* __restrict__ v_b,
              const float* __restrict__ o_w,  const float* __restrict__ o_b,
              float* __restrict__ out)
{
    // kfeat/vfeat: haloed features (bf16), padded rows => unguarded D stores
    __shared__ __align__(32) __bf16 kfeat[NROW][CKC];       // 21504 B
    __shared__ __align__(32) __bf16 vfeat[NROW][CVC];       // 21504 B
    // y in A-fragment layout: [mt(16)][lane(32)][16]
    __shared__ __align__(32) __bf16 yfrag[16*32*16];        // 16384 B
    // weights pre-swizzled into B-fragment layout:
    //  phase 1: [mat(2)][kc(8)][nt(2)][lane(32)][16]  (mat0=k1_w, mat1=v_w)
    //  phase 3 (reuse): [nt(16)][lane(32)][16]        (o_w)
    __shared__ __align__(32) __bf16 wfrag[16384];           // 32768 B

    const int tx   = blockIdx.x;
    const int ty   = blockIdx.y;
    const int b    = blockIdx.z;
    const int tid  = threadIdx.x;
    const int lane = tid & 31;
    const int wave = tid >> 5;             // 8 wave32
    const int ln16 = lane & 15;
    const int lh   = lane >> 4;

    const float* xb = x + (size_t)b * CH * HH * WW;

    // ===== Phase 0: stage k1_w / v_w into LDS, pre-swizzled to B-frag layout
    for (int i = tid; i < 16384; i += 256) {
        int e   = i & 15;
        int ln  = (i >> 4) & 31;
        int nt  = (i >> 9) & 1;
        int kc  = (i >> 10) & 7;
        int mat = (i >> 13) & 1;
        int c   = kc*32 + kmapB(ln >> 4, e >> 1) + (e & 1);
        int n   = nt*16 + (ln & 15);
        const float* w = mat ? v_w : k1_w;
        wfrag[i] = (__bf16)w[n*CH + c];
    }
    __syncthreads();

    // ===== Phase 1: k1 = relu(Wk x + b), v = Wv x + b on haloed 18x18 =======
    for (int mt = wave; mt < NMT; mt += 8) {
        v8f ak0 = {}; v8f ak1 = {}; v8f av0 = {}; v8f av1 = {};

        int pix = mt*16 + ln16;                 // this lane's A row
        if (pix >= NPIX) pix = NPIX - 1;        // clamp (dup loads, dead rows)
        int py = pix / HS, px = pix % HS;
        int gy = ty*TS - 1 + py, gx = tx*TS - 1 + px;
        int cgy = min(max(gy, 0), HH-1), cgx = min(max(gx, 0), WW-1);
        const float* xpix = xb + cgy*WW + cgx;

        for (int kc = 0; kc < CH/32; ++kc) {
            if (kc < CH/32 - 1)
                __builtin_prefetch(xpix + (size_t)(kc*32 + 32)*HH*WW, 0, 1);
            v16bf a;
            #pragma unroll
            for (int j = 0; j < 8; ++j) {
                int c0 = kc*32 + kmapA(lh, j);
                a[2*j+0] = (__bf16)xpix[(size_t)(c0+0)*HH*WW];
                a[2*j+1] = (__bf16)xpix[(size_t)(c0+1)*HH*WW];
            }
            const int base = ((kc*2)*32 + lane)*16;   // k1, nt0
            v16bf bk0 = *(const v16bf*)(&wfrag[base]);
            ak0 = __builtin_amdgcn_wmma_f32_16x16x32_bf16(false, a, false, bk0, (short)0, ak0, false, false);
            v16bf bk1 = *(const v16bf*)(&wfrag[base + 512]);
            ak1 = __builtin_amdgcn_wmma_f32_16x16x32_bf16(false, a, false, bk1, (short)0, ak1, false, false);
            v16bf bv0 = *(const v16bf*)(&wfrag[8192 + base]);
            av0 = __builtin_amdgcn_wmma_f32_16x16x32_bf16(false, a, false, bv0, (short)0, av0, false, false);
            v16bf bv1 = *(const v16bf*)(&wfrag[8192 + base + 512]);
            av1 = __builtin_amdgcn_wmma_f32_16x16x32_bf16(false, a, false, bv1, (short)0, av1, false, false);
        }

        // epilogue: bias (+relu for k), zero outside image (zero-pad semantics)
        float bk0s = k1_b[ln16], bk1s = k1_b[ln16+16];
        float bv0s = v_b[ln16],  bv1s = v_b[ln16+16];
        #pragma unroll
        for (int i = 0; i < 8; ++i) {
            int M = i + 8*lh;
            int p = mt*16 + M;                  // < NROW always: no guard
            int ppy = p / HS, ppx = p % HS;
            int ggy = ty*TS - 1 + ppy, ggx = tx*TS - 1 + ppx;
            bool inimg = (p < NPIX) & (ggy >= 0) & (ggy < HH) & (ggx >= 0) & (ggx < WW);
            kfeat[p][ln16]    = (__bf16)(inimg ? fmaxf(ak0[i] + bk0s, 0.f) : 0.f);
            kfeat[p][ln16+16] = (__bf16)(inimg ? fmaxf(ak1[i] + bk1s, 0.f) : 0.f);
            vfeat[p][ln16]    = (__bf16)(inimg ? (av0[i] + bv0s) : 0.f);
            vfeat[p][ln16+16] = (__bf16)(inimg ? (av1[i] + bv1s) : 0.f);
        }
    }
    __syncthreads();

    // ===== Phase 2a: restage o_w into LDS B-frag layout (wfrag is free now) ==
    for (int i = tid; i < 8192; i += 256) {
        int e  = i & 15;
        int ln = (i >> 4) & 31;
        int nt = (i >> 9) & 15;
        int cv = kmapB(ln >> 4, e >> 1) + (e & 1);
        int n  = nt*16 + (ln & 15);
        wfrag[i] = (__bf16)o_w[n*CVC + cv];
    }

    // ===== Phase 2b: depthwise 3x3 + CK->9 + softmax + value aggregation ====
    {
        const int p  = tid;                    // inner-tile pixel 0..255
        const int iy = p >> 4, ix = p & 15;

        float dwv[CKC];
        #pragma unroll
        for (int c = 0; c < CKC; ++c) dwv[c] = dw_b[c];
        #pragma unroll
        for (int di = 0; di < 3; ++di)
            #pragma unroll
            for (int dj = 0; dj < 3; ++dj) {
                int hp = (iy + di)*HS + (ix + dj);
                #pragma unroll
                for (int c = 0; c < CKC; ++c)
                    dwv[c] += (float)kfeat[hp][c] * dw_w[c*9 + di*3 + dj];
            }

        float logit[9], mx = -1e30f;
        #pragma unroll
        for (int t = 0; t < 9; ++t) {
            float s = k3_b[t];
            #pragma unroll
            for (int c = 0; c < CKC; ++c) s += k3_w[t*CKC + c] * dwv[c];
            logit[t] = s; mx = fmaxf(mx, s);
        }
        float den = 0.f;
        #pragma unroll
        for (int t = 0; t < 9; ++t) { logit[t] = __expf(logit[t] - mx); den += logit[t]; }
        float inv = 1.f / den;

        float acc[CVC];
        #pragma unroll
        for (int c = 0; c < CVC; ++c) acc[c] = 0.f;
        #pragma unroll
        for (int t = 0; t < 9; ++t) {
            int hp = (iy + t/3)*HS + (ix + t%3);
            float wgt = logit[t] * inv;
            #pragma unroll
            for (int c = 0; c < CVC; ++c) acc[c] += wgt * (float)vfeat[hp][c];
        }
        // scatter into A-fragment layout: pixel p = row (p&15) of mtile (p>>4)
        int mt2 = p >> 4, row = p & 15;
        #pragma unroll
        for (int c = 0; c < CVC; ++c) {
            int h, e; invA(c, h, e);
            yfrag[(mt2*32 + row + 16*h)*16 + e] = (__bf16)acc[c];
        }
    }
    __syncthreads();

    // ===== Phase 3: out = x + (y @ o_w^T + o_b), K = CV = 32 (one WMMA/tile) =
    for (int mt = wave; mt < 16; mt += 8) {
        v16bf a = *(const v16bf*)(&yfrag[(mt*32 + lane)*16]);
        const int gy = ty*TS + mt;
        for (int nt = 0; nt < 16; ++nt) {
            v16bf bb = *(const v16bf*)(&wfrag[(nt*32 + lane)*16]);
            v8f acc = {};
            acc = __builtin_amdgcn_wmma_f32_16x16x32_bf16(false, a, false, bb, (short)0, acc, false, false);
            int outc = nt*16 + ln16;
            float ob = o_b[outc];
            #pragma unroll
            for (int i = 0; i < 8; ++i) {
                int M  = i + 8*lh;
                int gx = tx*TS + M;
                size_t idx = (((size_t)b*CH + outc)*HH + gy)*WW + gx;
                out[idx] = acc[i] + ob + x[idx];
            }
        }
    }
}

extern "C" void kernel_launch(void* const* d_in, const int* in_sizes, int n_in,
                              void* d_out, int out_size, void* d_ws, size_t ws_size,
                              hipStream_t stream) {
    const float* x    = (const float*)d_in[0];
    const float* k1_w = (const float*)d_in[1];
    const float* k1_b = (const float*)d_in[2];
    const float* dw_w = (const float*)d_in[3];
    const float* dw_b = (const float*)d_in[4];
    const float* k3_w = (const float*)d_in[5];
    const float* k3_b = (const float*)d_in[6];
    const float* v_w  = (const float*)d_in[7];
    const float* v_b  = (const float*)d_in[8];
    const float* o_w  = (const float*)d_in[9];
    const float* o_b  = (const float*)d_in[10];
    float* out = (float*)d_out;

    dim3 grid(WW/TS, HH/TS, BATCH);   // 8 x 8 x 8 = 512 workgroups
    ns_fused<<<grid, 256, 0, stream>>>(x, k1_w, k1_b, dw_w, dw_b, k3_w, k3_b,
                                       v_w, v_b, o_w, o_b, out);
}
